// VectorQuantizer_30562987278584
// MI455X (gfx1250) — matrix operand order repaired and checked by reading, not compiled, not run on hardware
//
#include <hip/hip_runtime.h>
#include <hip/hip_bf16.h>
#include <math.h>

typedef __attribute__((ext_vector_type(16))) _Float16 v16h;
typedef __attribute__((ext_vector_type(8)))  float    v8f;

#define N_E_      16384
#define SPLIT0_   64
#define SPLIT1_   192
#define E_DIM_    256
#define TOK_      4096        // B*H*W = 4*32*32
#define BETA_     0.25f
#define ENT_RATIO_ 0.1f
#define INV_TEMP_ 100.0f      // 1/0.01
#define NT_TILES_ (N_E_/16)   // 1024 code tiles
#define MT_TILES_ (TOK_/16)   // 256 token tiles

// ---------------- workspace layout (byte offsets, all 256B aligned) ---------
#define OFF_ZNORM  ((size_t)0)                       // 4096*256 f32   = 4 MiB
#define OFF_ENORM  ((size_t)4*1024*1024)             // 16384*256 f32  = 16 MiB
#define OFF_AT     ((size_t)20*1024*1024)            // 4096*256 f16   = 2 MiB
#define OFF_BT     ((size_t)22*1024*1024)            // 16384*256 f16  = 8 MiB
#define OFF_ZSK    ((size_t)30*1024*1024)            // 4096 f32
#define OFF_ZSG    (OFF_ZSK + 16384)
#define OFF_ESK    (OFF_ZSG + 16384)                 // 16384 f32
#define OFF_ESG    (OFF_ESK + 65536)
#define OFF_RMAX   (OFF_ESG + 65536)                 // 4096 f32
#define OFF_RZ     (OFF_RMAX + 16384)
#define OFF_IDX    (OFF_RZ + 16384)                  // 4096 i32
#define OFF_AVGP   (OFF_IDX + 16384)                 // 16384 f32
#define OFF_ACC    (OFF_AVGP + 65536)                // 4 f32: ent, dk2, dg2, vq

// d_out layout: [0,1048576) z_q NCHW ; +0 vq ; +1 commit ; +2 entropy ;
//               +3 vqkd_d_norm ; +4 vqgan_d_norm ; [+5, +5+4096) idx
#define OUT_SCAL   1048576
#define OUT_IDX    (OUT_SCAL + 5)

// ---------------------------------------------------------------------------
__global__ __launch_bounds__(256) void vq_init(float* avgp, float* acc) {
  int i = blockIdx.x * 256 + threadIdx.x;
  if (i < N_E_) avgp[i] = 0.f;
  if (i < 4)    acc[i]  = 0.f;
}

// One block per token: normalize both splits, emit f32 row + WMMA-swizzled f16 A tile.
__global__ __launch_bounds__(256) void vq_prep_tokens(const float* __restrict__ z,
    float* __restrict__ znorm, _Float16* __restrict__ at,
    float* __restrict__ zsk, float* __restrict__ zsg) {
  int t = blockIdx.x, c = threadIdx.x;
  int b = t >> 10, hw = t & 1023;
  float v = z[(size_t)b*262144 + (size_t)c*1024 + (size_t)hw];
  __shared__ float sk, sg;
  if (c == 0) { sk = 0.f; sg = 0.f; }
  __syncthreads();
  atomicAdd(c < SPLIT0_ ? &sk : &sg, v * v);
  __syncthreads();
  float s = (c < SPLIT0_) ? sk : sg;
  float nrm = fmaxf(sqrtf(s), 1e-12f);
  float vn = v / nrm;
  znorm[(size_t)t * E_DIM_ + c] = vn;
  // A-matrix f16 16x32 per-lane layout: lane = m + 16*k[3], elem = (k&7) + 8*k[4]
  int mt = t >> 4, m = t & 15;
  int kb = c >> 5, kk = c & 31;
  int lane = m + (((kk >> 3) & 1) << 4);
  int h    = (kk & 7) + ((kk >> 4) << 3);
  at[(((size_t)mt * 8 + kb) * 32 + lane) * 16 + h] = (_Float16)vn;
  if (c == 0) {
    float nk = fmaxf(sqrtf(sk), 1e-12f), ng = fmaxf(sqrtf(sg), 1e-12f);
    zsk[t] = sk / (nk * nk);
    zsg[t] = sg / (ng * ng);
  }
}

// One block per code: normalize, emit f32 row + WMMA-swizzled f16 B tile.
__global__ __launch_bounds__(256) void vq_prep_codes(const float* __restrict__ wk,
    const float* __restrict__ wg, float* __restrict__ enorm, _Float16* __restrict__ bt,
    float* __restrict__ esk, float* __restrict__ esg) {
  int n = blockIdx.x, c = threadIdx.x;
  float v = (c < SPLIT0_) ? wk[(size_t)n * SPLIT0_ + c]
                          : wg[(size_t)n * SPLIT1_ + (c - SPLIT0_)];
  __shared__ float sk, sg;
  if (c == 0) { sk = 0.f; sg = 0.f; }
  __syncthreads();
  atomicAdd(c < SPLIT0_ ? &sk : &sg, v * v);
  __syncthreads();
  float s = (c < SPLIT0_) ? sk : sg;
  float nrm = fmaxf(sqrtf(s), 1e-12f);
  float vn = v / nrm;
  enorm[(size_t)n * E_DIM_ + c] = vn;
  // B-matrix f16 32x16 per-lane layout: lane = n + 16*k[3], elem = (k&7) + 8*k[4]
  int nt = n >> 4, nn = n & 15;
  int kb = c >> 5, kk = c & 31;
  int lane = nn + (((kk >> 3) & 1) << 4);
  int h    = (kk & 7) + ((kk >> 4) << 3);
  bt[(((size_t)nt * 8 + kb) * 32 + lane) * 16 + h] = (_Float16)vn;
  if (c == 0) {
    float nk = fmaxf(sqrtf(sk), 1e-12f), ng = fmaxf(sqrtf(sg), 1e-12f);
    esk[n] = sk / (nk * nk);
    esg[n] = sg / (ng * ng);
  }
}

// Pass 1: per 16-token tile, sweep all 16384 codes with WMMA; lane-local online
// softmax / argmin / sum-d^2, merged cross-lane at the end.
__global__ __launch_bounds__(128) void vq_pass1(const _Float16* __restrict__ at,
    const _Float16* __restrict__ bt, const float* __restrict__ zsk,
    const float* __restrict__ zsg, const float* __restrict__ esk,
    const float* __restrict__ esg, float* __restrict__ rmax, float* __restrict__ rz,
    int* __restrict__ idxw, float* __restrict__ acc, float* __restrict__ out) {
  int lane = threadIdx.x & 31;
  int mt = blockIdx.x * 4 + (threadIdx.x >> 5);
  int ncol = lane & 15;

  v16h a[8];
#pragma unroll
  for (int kb = 0; kb < 8; ++kb)
    a[kb] = *(const v16h*)(at + (((size_t)mt * 8 + kb) * 32 + lane) * 16);

  float zk2[8], zg2[8];
#pragma unroll
  for (int r = 0; r < 8; ++r) {
    int m = mt * 16 + r + ((lane >> 4) << 3);
    zk2[r] = zsk[m]; zg2[r] = zsg[m];
  }

  float dmin[8], M[8], S0[8], S1[8], dk2a[8], dg2a[8];
  int imin[8];
#pragma unroll
  for (int r = 0; r < 8; ++r) {
    dmin[r] = 1e30f; M[r] = -1e30f; S0[r] = 0.f; S1[r] = 0.f;
    dk2a[r] = 0.f; dg2a[r] = 0.f; imin[r] = 0;
  }

  for (int nt = 0; nt < NT_TILES_; ++nt) {
    v8f ck = {}; v8f cg = {};
#pragma unroll
    for (int kb = 0; kb < 8; ++kb) {
      v16h b = *(const v16h*)(bt + (((size_t)nt * 8 + kb) * 32 + lane) * 16);
      if (kb < 2)
        ck = __builtin_amdgcn_wmma_f32_16x16x32_f16(false, a[kb], false, b,
                                                    (short)0, ck, false, false);
      else
        cg = __builtin_amdgcn_wmma_f32_16x16x32_f16(false, a[kb], false, b,
                                                    (short)0, cg, false, false);
    }
    int ncode = nt * 16 + ncol;
    float ek2 = esk[ncode], eg2 = esg[ncode];
#pragma unroll
    for (int r = 0; r < 8; ++r) {
      float dk = zk2[r] + ek2 - 2.f * ck[r];
      float dg = zg2[r] + eg2 - 2.f * cg[r];
      float d  = dk + dg;
      dk2a[r] += dk * dk;
      dg2a[r] += dg * dg;
      if (d < dmin[r]) { dmin[r] = d; imin[r] = ncode; }
      float aV = -INV_TEMP_ * d;
      float nM = fmaxf(M[r], aV);
      float sc = __expf(M[r] - nM);
      float e  = __expf(aV - nM);
      S1[r] = sc * (S1[r] + (M[r] - nM) * S0[r]) + (aV - nM) * e;
      S0[r] = sc * S0[r] + e;
      M[r]  = nM;
    }
  }

#pragma unroll
  for (int r = 0; r < 8; ++r) {
#pragma unroll
    for (int msk = 1; msk < 16; msk <<= 1) {
      float od = __shfl_xor(dmin[r], msk, 32);
      int   oi = __shfl_xor(imin[r], msk, 32);
      if (od < dmin[r] || (od == dmin[r] && oi < imin[r])) { dmin[r] = od; imin[r] = oi; }
      float oM = __shfl_xor(M[r],  msk, 32);
      float o0 = __shfl_xor(S0[r], msk, 32);
      float o1 = __shfl_xor(S1[r], msk, 32);
      float nM = fmaxf(M[r], oM);
      float s1 = __expf(M[r] - nM), s2 = __expf(oM - nM);
      S1[r] = s1 * (S1[r] + (M[r] - nM) * S0[r]) + s2 * (o1 + (oM - nM) * o0);
      S0[r] = s1 * S0[r] + s2 * o0;
      M[r]  = nM;
      dk2a[r] += __shfl_xor(dk2a[r], msk, 32);
      dg2a[r] += __shfl_xor(dg2a[r], msk, 32);
    }
    if ((lane & 15) == 0) {
      int m = mt * 16 + r + ((lane >> 4) << 3);
      rmax[m] = M[r];
      rz[m]   = S0[r];
      idxw[m] = imin[r];
      out[OUT_IDX + m] = (float)imin[r];
      float ent = logf(S0[r]) - S1[r] / S0[r];   // -sum p log p for this row
      atomicAdd(&acc[0], ent);
      atomicAdd(&acc[1], dk2a[r]);
      atomicAdd(&acc[2], dg2a[r]);
    }
  }
}

// Pass 2: recompute d tiles, accumulate avg_probs[n] (codebook is L2-resident).
__global__ __launch_bounds__(128) void vq_pass2(const _Float16* __restrict__ at,
    const _Float16* __restrict__ bt, const float* __restrict__ zsk,
    const float* __restrict__ zsg, const float* __restrict__ esk,
    const float* __restrict__ esg, const float* __restrict__ rmax,
    const float* __restrict__ rz, float* __restrict__ avgp) {
  int lane = threadIdx.x & 31;
  int mt = blockIdx.x * 4 + (threadIdx.x >> 5);
  int ncol = lane & 15;

  v16h a[8];
#pragma unroll
  for (int kb = 0; kb < 8; ++kb)
    a[kb] = *(const v16h*)(at + (((size_t)mt * 8 + kb) * 32 + lane) * 16);

  float zk2[8], zg2[8], rM[8], rIZ[8];
#pragma unroll
  for (int r = 0; r < 8; ++r) {
    int m = mt * 16 + r + ((lane >> 4) << 3);
    zk2[r] = zsk[m]; zg2[r] = zsg[m];
    rM[r] = rmax[m]; rIZ[r] = 1.f / rz[m];
  }

  for (int nt = 0; nt < NT_TILES_; ++nt) {
    v8f ck = {}; v8f cg = {};
#pragma unroll
    for (int kb = 0; kb < 8; ++kb) {
      v16h b = *(const v16h*)(bt + (((size_t)nt * 8 + kb) * 32 + lane) * 16);
      if (kb < 2)
        ck = __builtin_amdgcn_wmma_f32_16x16x32_f16(false, a[kb], false, b,
                                                    (short)0, ck, false, false);
      else
        cg = __builtin_amdgcn_wmma_f32_16x16x32_f16(false, a[kb], false, b,
                                                    (short)0, cg, false, false);
    }
    int ncode = nt * 16 + ncol;
    float ek2 = esk[ncode], eg2 = esg[ncode];
    float s = 0.f;
#pragma unroll
    for (int r = 0; r < 8; ++r) {
      float dk = zk2[r] + ek2 - 2.f * ck[r];
      float dg = zg2[r] + eg2 - 2.f * cg[r];
      float aV = -INV_TEMP_ * (dk + dg);
      s += __expf(aV - rM[r]) * rIZ[r];
    }
    atomicAdd(&avgp[ncode], s);
  }
}

// avg-entropy reduction + entropy_loss + d-norm scalars.
__global__ __launch_bounds__(256) void vq_entropy_final(const float* __restrict__ avgp,
    const float* __restrict__ acc, float* __restrict__ out) {
  __shared__ float ssum;
  if (threadIdx.x == 0) ssum = 0.f;
  __syncthreads();
  float local = 0.f;
  for (int i = threadIdx.x; i < N_E_; i += 256) {
    float ap = avgp[i] * (1.0f / TOK_);
    local += -ap * logf(ap + 1e-5f);
  }
  atomicAdd(&ssum, local);
  __syncthreads();
  if (threadIdx.x == 0) {
    float avg_ent = ssum;
    float samp    = acc[0] * (1.0f / TOK_);
    out[OUT_SCAL + 2] = ENT_RATIO_ * (samp - avg_ent);
    out[OUT_SCAL + 3] = acc[1] * (1.0f / TOK_);
    out[OUT_SCAL + 4] = acc[2] * (1.0f / TOK_);
  }
}

// Gather z_q (== straight-through value), write NCHW, accumulate MSE.
__global__ __launch_bounds__(256) void vq_gather(const float* __restrict__ enorm,
    const float* __restrict__ znorm, const int* __restrict__ idxw,
    float* __restrict__ out, float* __restrict__ acc) {
  int t = blockIdx.x, c = threadIdx.x;
  int id = idxw[t];
  float q  = enorm[(size_t)id * E_DIM_ + c];
  float zc = znorm[(size_t)t * E_DIM_ + c];
  float diff = q - zc;
  __shared__ float s;
  if (c == 0) s = 0.f;
  __syncthreads();
  atomicAdd(&s, diff * diff);
  __syncthreads();
  int b = t >> 10, hw = t & 1023;
  out[(size_t)b * 262144 + (size_t)c * 1024 + hw] = q;
  if (c == 0) atomicAdd(&acc[3], s);
}

__global__ void vq_final(const float* __restrict__ acc, float* __restrict__ out) {
  float vq = acc[3] * (1.0f / ((float)TOK_ * (float)E_DIM_));
  out[OUT_SCAL + 0] = vq;
  out[OUT_SCAL + 1] = BETA_ * vq;
}

// ---------------------------------------------------------------------------
extern "C" void kernel_launch(void* const* d_in, const int* in_sizes, int n_in,
                              void* d_out, int out_size, void* d_ws, size_t ws_size,
                              hipStream_t stream) {
  const float* z  = (const float*)d_in[0];
  const float* wk = (const float*)d_in[1];
  const float* wg = (const float*)d_in[2];
  float* out = (float*)d_out;
  char* ws = (char*)d_ws;

  float*    znorm = (float*)   (ws + OFF_ZNORM);
  float*    enorm = (float*)   (ws + OFF_ENORM);
  _Float16* at    = (_Float16*)(ws + OFF_AT);
  _Float16* bt    = (_Float16*)(ws + OFF_BT);
  float*    zsk   = (float*)   (ws + OFF_ZSK);
  float*    zsg   = (float*)   (ws + OFF_ZSG);
  float*    esk   = (float*)   (ws + OFF_ESK);
  float*    esg   = (float*)   (ws + OFF_ESG);
  float*    rmax  = (float*)   (ws + OFF_RMAX);
  float*    rz    = (float*)   (ws + OFF_RZ);
  int*      idxw  = (int*)     (ws + OFF_IDX);
  float*    avgp  = (float*)   (ws + OFF_AVGP);
  float*    acc   = (float*)   (ws + OFF_ACC);

  vq_init        <<<(N_E_ + 255) / 256, 256, 0, stream>>>(avgp, acc);
  vq_prep_tokens <<<TOK_,  256, 0, stream>>>(z, znorm, at, zsk, zsg);
  vq_prep_codes  <<<N_E_,  256, 0, stream>>>(wk, wg, enorm, bt, esk, esg);
  vq_pass1       <<<MT_TILES_ / 4, 128, 0, stream>>>(at, bt, zsk, zsg, esk, esg,
                                                     rmax, rz, idxw, acc, out);
  vq_pass2       <<<MT_TILES_ / 4, 128, 0, stream>>>(at, bt, zsk, zsg, esk, esg,
                                                     rmax, rz, avgp);
  vq_entropy_final<<<1, 256, 0, stream>>>(avgp, acc, out);
  vq_gather      <<<TOK_, 256, 0, stream>>>(enorm, znorm, idxw, out, acc);
  vq_final       <<<1, 1, 0, stream>>>(acc, out);
}
